// HeteroGLSTM_34256659152988
// MI455X (gfx1250) — matrix operand order repaired and checked by reading, not compiled, not run on hardware
//
#include <hip/hip_runtime.h>
#include <hip/hip_bf16.h>
#include <math.h>

// ---------------------------------------------------------------------------
// HeteroGLSTM on MI455X (gfx1250), wave32 + WMMA bf16 split-precision GEMM.
//
// Exact simplifications vs reference:
//  * f-gate multiplies the zero initial cell state -> skipped entirely.
//  * layer-0 seg_mean aggregation is gate-invariant -> computed once.
//  * degree counts computed once; 1/deg folded into GEMM A-row loads.
// GEMMs: f32 emulated as bf16 hi/lo (3 WMMA products) on v_wmma_f32_16x16x32_bf16.
// ---------------------------------------------------------------------------

#define NM_NODES 50000
#define NJ_NODES 50000
#define N_EDGES  600000
#define CH       128

typedef __bf16 bf16_t;
typedef __attribute__((ext_vector_type(16))) __bf16 v16bf;
typedef __attribute__((ext_vector_type(8)))  float  v8f;

__device__ __forceinline__ v8f wmma_bf16(v16bf a, v16bf b, v8f c) {
  // (neg_a, A, neg_b, B, c_mod, C, reuse_a, reuse_b)
  return __builtin_amdgcn_wmma_f32_16x16x32_bf16(false, a, false, b, (short)0, c, false, false);
}

// A fragment: 16x32 bf16 tile from row-major f32 [M][128], converted to hi/lo
// bf16 in contiguous k order matching B: lane&15 = M row; lanes>=16 take K+16;
// VGPR v in 0..7 -> K = 2v..2v+1 (per half-lane group, contiguous like B).
__device__ __forceinline__ void load_a_frag(const float* A, int row0, int k0,
                                            const float* rowScale,
                                            v16bf& ah, v16bf& al) {
  const int l  = threadIdx.x & 31;
  const int m  = row0 + (l & 15);
  const int kb = k0 + ((l >> 4) << 4);
  const float rs = rowScale ? rowScale[m] : 1.0f;
  const float* Ar = A + (size_t)m * CH;
#pragma unroll
  for (int v = 0; v < 8; ++v) {
    const int k = kb + (v << 1);
    const float2 f2 = *(const float2*)(Ar + k);
    const float f0 = f2.x * rs, f1 = f2.y * rs;
    const __bf16 h0 = (__bf16)f0, h1 = (__bf16)f1;
    ah[2 * v]     = h0;
    ah[2 * v + 1] = h1;
    al[2 * v]     = (__bf16)(f0 - (float)h0);
    al[2 * v + 1] = (__bf16)(f1 - (float)h1);
  }
}

// B fragment: 32x16 bf16 tile from pre-transposed weights Wt[n][k] (ld=128).
// Lane&15 = N column; lanes>=16 take K+16; per-lane data = 16 contiguous bf16.
__device__ __forceinline__ v16bf load_b_frag(const bf16_t* __restrict__ Wt,
                                             int k0, int n0) {
  const int l = threadIdx.x & 31;
  const bf16_t* p = Wt + (size_t)(n0 + (l & 15)) * CH + k0 + ((l >> 4) << 4);
  return *(const v16bf*)p;  // 32B aligned, one wide load per lane
}

__device__ __forceinline__ float apply_act(float v, int act) {
  if (act == 1) return 1.0f / (1.0f + expf(-v));  // sigmoid
  if (act == 2) return tanhf(v);
  return v;
}

// Fused SAGE update: Out = act( (diag(aggScale)*Aagg) @ Wl + bias + Ax @ Wr )
// One wave -> 16 rows x 128 cols. 8 waves per block. In-place Out==Ax is safe
// (each row is read only by the wave that writes it, stores after all loads).
__global__ __launch_bounds__(256) void gemm_sage(
    const float* Aagg, const float* aggScale, const float* Ax,
    const bf16_t* __restrict__ Wlh, const bf16_t* __restrict__ Wll,
    const bf16_t* __restrict__ Wrh, const bf16_t* __restrict__ Wrl,
    const float* __restrict__ bias, float* Out, int M, int act) {
  const int wave = threadIdx.x >> 5;
  const int lane = threadIdx.x & 31;
  const int row0 = (blockIdx.x * 8 + wave) * 16;
  if (row0 >= M) return;  // wave-uniform exit: EXEC stays all-1s for WMMA

  const int col = lane & 15;
  v8f acc[8];
#pragma unroll
  for (int n = 0; n < 8; ++n) {
    const float bv = bias[(n << 4) + col];
#pragma unroll
    for (int r = 0; r < 8; ++r) acc[n][r] = bv;
  }

  for (int k0 = 0; k0 < CH; k0 += 32) {
    v16bf ah, al;
    load_a_frag(Aagg, row0, k0, aggScale, ah, al);
#pragma unroll
    for (int n = 0; n < 8; ++n) {
      const v16bf bh = load_b_frag(Wlh, k0, n << 4);
      const v16bf bl = load_b_frag(Wll, k0, n << 4);
      acc[n] = wmma_bf16(ah, bh, acc[n]);
      acc[n] = wmma_bf16(al, bh, acc[n]);
      acc[n] = wmma_bf16(ah, bl, acc[n]);
    }
    load_a_frag(Ax, row0, k0, nullptr, ah, al);
#pragma unroll
    for (int n = 0; n < 8; ++n) {
      const v16bf bh = load_b_frag(Wrh, k0, n << 4);
      const v16bf bl = load_b_frag(Wrl, k0, n << 4);
      acc[n] = wmma_bf16(ah, bh, acc[n]);
      acc[n] = wmma_bf16(al, bh, acc[n]);
      acc[n] = wmma_bf16(ah, bl, acc[n]);
    }
  }

  // C/D layout: VGPR r, lanes<16 -> row r ; lanes>=16 -> row r+8 ; col = lane&15.
  const int rrow = row0 + ((lane >> 4) << 3);
#pragma unroll
  for (int n = 0; n < 8; ++n)
#pragma unroll
    for (int r = 0; r < 8; ++r)
      Out[(size_t)(rrow + r) * CH + (n << 4) + col] = apply_act(acc[n][r], act);
}

// --- weight prep: f32 [mats][k][n] -> transposed bf16 hi/lo [mats][n][k] ----
__global__ void cvt_weights(const float* __restrict__ W, bf16_t* __restrict__ Wh,
                            bf16_t* __restrict__ Wl, int total) {
  const int i = blockIdx.x * blockDim.x + threadIdx.x;
  if (i >= total) return;
  const int mat = i >> 14, rc = i & 16383, k = rc >> 7, n = rc & 127;
  const float f = W[i];
  const __bf16 h = (__bf16)f;
  const int o = (mat << 14) | (n << 7) | k;
  Wh[o] = h;
  Wl[o] = (__bf16)(f - (float)h);
}

// --- aggregation helpers ----------------------------------------------------
__global__ void fill_zero(float* __restrict__ p, size_t n) {
  size_t i = (size_t)blockIdx.x * blockDim.x + threadIdx.x;
  const size_t stride = (size_t)gridDim.x * blockDim.x;
  for (; i < n; i += stride) p[i] = 0.0f;
}

__global__ void count_deg(const int* __restrict__ dst, float* __restrict__ deg, int E) {
  const int e = blockIdx.x * blockDim.x + threadIdx.x;
  if (e < E) atomicAdd(&deg[dst[e]], 1.0f);
}

__global__ void recip_deg(float* __restrict__ d, int n) {
  const int i = blockIdx.x * blockDim.x + threadIdx.x;
  if (i < n) d[i] = 1.0f / fmaxf(d[i], 1.0f);
}

// 32 lanes per edge: coalesced 512B row gather + float atomics scatter.
__global__ __launch_bounds__(256) void scatter_add(
    const float* __restrict__ X, const int* __restrict__ src,
    const int* __restrict__ dst, float* __restrict__ Agg, int E) {
  const int t = blockIdx.x * blockDim.x + threadIdx.x;
  const int e = t >> 5;
  if (e >= E) return;
  const int c = (t & 31) << 2;
  const float4 v = *(const float4*)(X + (size_t)src[e] * CH + c);
  float* p = Agg + (size_t)dst[e] * CH + c;
  atomicAdd(p + 0, v.x);
  atomicAdd(p + 1, v.y);
  atomicAdd(p + 2, v.z);
  atomicAdd(p + 3, v.w);
}

// --- final LSTM combine: c = i*t ; h = o*tanh(c) ----------------------------
__global__ void lstm_combine(const float* __restrict__ gi, const float* __restrict__ gt,
                             const float* __restrict__ go, float* __restrict__ h,
                             float* __restrict__ c, size_t n) {
  size_t i = (size_t)blockIdx.x * blockDim.x + threadIdx.x;
  const size_t stride = (size_t)gridDim.x * blockDim.x;
  for (; i < n; i += stride) {
    const float cv = gi[i] * gt[i];
    c[i] = cv;
    h[i] = go[i] * tanhf(cv);
  }
}

extern "C" void kernel_launch(void* const* d_in, const int* in_sizes, int n_in,
                              void* d_out, int out_size, void* d_ws, size_t ws_size,
                              hipStream_t stream) {
  const float* x_m  = (const float*)d_in[0];
  const float* x_j  = (const float*)d_in[1];
  const float* Wl   = (const float*)d_in[2];  // [G][L][2][C][C]
  const float* Wr   = (const float*)d_in[3];
  const float* bias = (const float*)d_in[4];  // [G][L][2][C]
  const int*   ei_mj = (const int*)d_in[5];   // [2][E]: row0 src(m), row1 dst(j)
  const int*   ei_jm = (const int*)d_in[6];
  float* out = (float*)d_out;                 // [4][N][C]: h_m, h_j, c_m, c_j

  const int NM = NM_NODES, NJ = NJ_NODES, E = N_EDGES;
  const size_t NC = (size_t)NM * CH;

  // workspace carving (256B aligned)
  char* w = (char*)d_ws;
  auto carve = [&](size_t bytes) -> char* {
    char* p = w;
    w += (bytes + 255) & ~(size_t)255;
    return p;
  };
  float* deg_j = (float*)carve((size_t)NJ * 4);
  float* deg_m = (float*)carve((size_t)NM * 4);
  float* agg_j = (float*)carve(NC * 4);
  float* agg_m = (float*)carve(NC * 4);
  float* xm_g[3];
  float* xj_g[3];
  for (int s = 0; s < 3; ++s) {
    xm_g[s] = (float*)carve(NC * 4);
    xj_g[s] = (float*)carve(NC * 4);
  }
  const size_t WTOT = (size_t)4 * 2 * 2 * CH * CH;  // G*L*2*C*C
  bf16_t* Wlh = (bf16_t*)carve(WTOT * 2);
  bf16_t* Wll = (bf16_t*)carve(WTOT * 2);
  bf16_t* Wrh = (bf16_t*)carve(WTOT * 2);
  bf16_t* Wrl = (bf16_t*)carve(WTOT * 2);

  // 1) weights -> transposed bf16 hi/lo
  cvt_weights<<<(int)((WTOT + 255) / 256), 256, 0, stream>>>(Wl, Wlh, Wll, (int)WTOT);
  cvt_weights<<<(int)((WTOT + 255) / 256), 256, 0, stream>>>(Wr, Wrh, Wrl, (int)WTOT);

  // 2) degrees (gate/layer invariant), then 1/max(deg,1)
  fill_zero<<<256, 256, 0, stream>>>(deg_j, (size_t)NJ);
  fill_zero<<<256, 256, 0, stream>>>(deg_m, (size_t)NM);
  count_deg<<<(E + 255) / 256, 256, 0, stream>>>(ei_mj + E, deg_j, E);
  count_deg<<<(E + 255) / 256, 256, 0, stream>>>(ei_jm + E, deg_m, E);
  recip_deg<<<(NJ + 255) / 256, 256, 0, stream>>>(deg_j, NJ);
  recip_deg<<<(NM + 255) / 256, 256, 0, stream>>>(deg_m, NM);

  auto agg_pass = [&](const float* feat, const int* ei, float* agg, int Ndst) {
    fill_zero<<<1024, 256, 0, stream>>>(agg, (size_t)Ndst * CH);
    scatter_add<<<(E * 32 + 255) / 256, 256, 0, stream>>>(feat, ei, ei + E, agg, E);
  };

  const int gates[3] = {0, 2, 3};   // i, c(tanh), o — f-gate is dead (c0==0)
  const int acts[3]  = {1, 2, 1};   // sigmoid, tanh, sigmoid
  auto woff = [&](int g, int l, int t) { return (size_t)((g * 2 + l) * 2 + t) * CH * CH; };
  auto boff = [&](int g, int l, int t) { return (size_t)((g * 2 + l) * 2 + t) * CH; };
  const int gblocks = ((NM / 16) + 7) / 8;  // 3125 strips / 8 waves

  // 3) layer 0: aggregation shared across gates, GEMM per gate
  agg_pass(x_m, ei_mj, agg_j, NJ);
  agg_pass(x_j, ei_jm, agg_m, NM);
  for (int s = 0; s < 3; ++s) {
    const int g = gates[s];
    gemm_sage<<<gblocks, 256, 0, stream>>>(
        agg_j, deg_j, x_j, Wlh + woff(g, 0, 0), Wll + woff(g, 0, 0),
        Wrh + woff(g, 0, 0), Wrl + woff(g, 0, 0), bias + boff(g, 0, 0),
        xj_g[s], NJ, 0);
    gemm_sage<<<gblocks, 256, 0, stream>>>(
        agg_m, deg_m, x_m, Wlh + woff(g, 0, 1), Wll + woff(g, 0, 1),
        Wrh + woff(g, 0, 1), Wrl + woff(g, 0, 1), bias + boff(g, 0, 1),
        xm_g[s], NM, 0);
  }

  // 4) layer 1 per gate: aggregate layer-0 outputs, fused GEMM + activation
  //    (in-place over the root input; safe per-wave read-then-write)
  for (int s = 0; s < 3; ++s) {
    const int g = gates[s];
    agg_pass(xm_g[s], ei_mj, agg_j, NJ);
    agg_pass(xj_g[s], ei_jm, agg_m, NM);
    gemm_sage<<<gblocks, 256, 0, stream>>>(
        agg_j, deg_j, xj_g[s], Wlh + woff(g, 1, 0), Wll + woff(g, 1, 0),
        Wrh + woff(g, 1, 0), Wrl + woff(g, 1, 0), bias + boff(g, 1, 0),
        xj_g[s], NJ, acts[s]);
    gemm_sage<<<gblocks, 256, 0, stream>>>(
        agg_m, deg_m, xm_g[s], Wlh + woff(g, 1, 1), Wll + woff(g, 1, 1),
        Wrh + woff(g, 1, 1), Wrl + woff(g, 1, 1), bias + boff(g, 1, 1),
        xm_g[s], NM, acts[s]);
  }

  // 5) combine: out = [h_m, h_j, c_m, c_j]
  lstm_combine<<<1024, 256, 0, stream>>>(xm_g[0], xm_g[1], xm_g[2],
                                         out + 0 * NC, out + 2 * NC, NC);
  lstm_combine<<<1024, 256, 0, stream>>>(xj_g[0], xj_g[1], xj_g[2],
                                         out + 1 * NC, out + 3 * NC, NC);
}
